// MultiHeadAttention_39453569581733
// MI455X (gfx1250) — matrix operand order repaired and checked by reading, not compiled, not run on hardware
//
#include <hip/hip_runtime.h>
#include <hip/hip_bf16.h>
#include <stdint.h>

typedef __attribute__((ext_vector_type(16))) __bf16   v16bf;
typedef __attribute__((ext_vector_type(8)))  float    v8f;
typedef __attribute__((ext_vector_type(8)))  uint32_t v8u;

__device__ __forceinline__ uint16_t f32_to_bf16_bits(float f) {
    uint32_t x = __builtin_bit_cast(uint32_t, f);
    uint32_t r = (x + 0x7FFFu + ((x >> 16) & 1u)) >> 16;
    return (uint16_t)r;
}

// A operand, k order used here (not ISA table): lane (g = lane/16, m = lane%16), VGPR v holds
// bf16 pair at K = kA_idx(v,g) = g*16 + v*2, kA_idx+1 (same order as B)
__device__ __forceinline__ int kA_idx(int v, int g) {
    return g * 16 + v * 2;
}
// B operand, k order used here (matches A): lane (g, n = lane%16), VGPR v holds pair at
// K = g*16 + v*2

// Async global -> LDS copy, 8 bytes per lane (CDNA5 GLOBAL_LOAD_ASYNC_TO_LDS_B64,
// tracked by ASYNCcnt).
__device__ __forceinline__ void async_ld_b64(void* lds, const void* gptr) {
    asm volatile("global_load_async_to_lds_b64 %0, %1, off"
                 :: "v"((uint32_t)(uintptr_t)lds), "v"(gptr)
                 : "memory");
}
#define ASYNC_WAIT(n) asm volatile("s_wait_asynccnt " #n ::: "memory")

// ---------------------------------------------------------------------------
// Kernel 0: prep — convert x to bf16, pre-transpose W_qkv / W_out to bf16
// [N][K] so GEMM B tiles are row-contiguous for async LDS staging.
// ---------------------------------------------------------------------------
__global__ __launch_bounds__(256) void prep_kernel(
    const float* __restrict__ x, const float* __restrict__ Wqkv,
    const float* __restrict__ Wout, uint16_t* __restrict__ xb,
    uint16_t* __restrict__ wqkvT, uint16_t* __restrict__ woutT)
{
    const int i = blockIdx.x * 256 + threadIdx.x;
    if (i < 4194304) {                                  // x: [8192*512] f32
        xb[i] = f32_to_bf16_bits(x[i]);
    } else if (i < 4194304 + 786432) {                  // W_qkv: [512][1536]
        const int j = i - 4194304;
        const int k = j / 1536, n = j % 1536;
        wqkvT[(size_t)n * 512 + k] = f32_to_bf16_bits(Wqkv[j]);
    } else if (i < 4194304 + 786432 + 262144) {         // W_out: [512][512]
        const int j = i - (4194304 + 786432);
        const int k = j >> 9, n = j & 511;
        woutT[(size_t)n * 512 + k] = f32_to_bf16_bits(Wout[j]);
    }
}

// Double-buffered async staging of a 128x32 bf16 tile pair (A rows from Ab,
// B rows from Bt, both K-contiguous with row stride 512), 8 B64 ops/thread.
#define GSTAGE(bufi, kk_)                                                        \
    do {                                                                         \
        const int _kk = (kk_);                                                   \
        _Pragma("unroll")                                                        \
        for (int _p = 0; _p < 4; ++_p) {                                         \
            const int _i = _p * 256 + tid;                                       \
            const int _r = _i >> 3, _c = _i & 7;                                 \
            async_ld_b64(&sA[bufi][_r][_c * 4],                                  \
                         &Ab[(size_t)(blockM + _r) * 512 + _kk + _c * 4]);       \
        }                                                                        \
        _Pragma("unroll")                                                        \
        for (int _p = 0; _p < 4; ++_p) {                                         \
            const int _i = _p * 256 + tid;                                       \
            const int _r = _i >> 3, _c = _i & 7;                                 \
            async_ld_b64(&sB[bufi][_r][_c * 4],                                  \
                         &Bt[(size_t)(blockN + _r) * 512 + _kk + _c * 4]);       \
        }                                                                        \
    } while (0)

// Shared GEMM main loop: 128x128 block, 8 waves (wave = 32x64), K = 512.
// Leaves results in acc[2][4]; all-bf16 inputs, async double-buffered LDS.
#define GEMM_MAIN_LOOP()                                                         \
    GSTAGE(0, 0);                                                                \
    for (int ks = 0; ks < 16; ++ks) {                                            \
        const int buf = ks & 1;                                                  \
        if (ks < 15) { GSTAGE(buf ^ 1, (ks + 1) * 32); ASYNC_WAIT(0x8); }        \
        else         { ASYNC_WAIT(0x0); }                                        \
        __syncthreads();                                                         \
        v16bf a[2];                                                              \
        _Pragma("unroll")                                                        \
        for (int mt = 0; mt < 2; ++mt) {                                         \
            v8u au;                                                              \
            _Pragma("unroll")                                                    \
            for (int v = 0; v < 8; ++v)                                          \
                au[v] = *(const uint32_t*)&sA[buf][waveM + mt * 16 + ln16][kA_idx(v, g)]; \
            a[mt] = __builtin_bit_cast(v16bf, au);                               \
        }                                                                        \
        _Pragma("unroll")                                                        \
        for (int nt = 0; nt < 4; ++nt) {                                         \
            v8u bu;                                                              \
            _Pragma("unroll")                                                    \
            for (int v = 0; v < 8; ++v)                                          \
                bu[v] = *(const uint32_t*)&sB[buf][waveN + nt * 16 + ln16][g * 16 + v * 2]; \
            v16bf bfr = __builtin_bit_cast(v16bf, bu);                           \
            _Pragma("unroll")                                                    \
            for (int mt = 0; mt < 2; ++mt)                                       \
                acc[mt][nt] = __builtin_amdgcn_wmma_f32_16x16x32_bf16(           \
                    false, a[mt], false, bfr, (short)0, acc[mt][nt], false, false); \
        }                                                                        \
        __syncthreads();                                                         \
    }

// ---------------------------------------------------------------------------
// Kernel 1: qkv = xb @ W_qkv^T(pre-transposed) + b_qkv, scattered to bf16
// q[B,H,N,hd], k[B,H,N,hd], v TRANSPOSED [B,H,hd,N].
// M = 8192, K = 512, N = 1536.
// ---------------------------------------------------------------------------
__global__ __launch_bounds__(256) void qkv_gemm_kernel(
    const uint16_t* __restrict__ Ab, const uint16_t* __restrict__ Bt,
    const float* __restrict__ bqkv, uint16_t* __restrict__ qb,
    uint16_t* __restrict__ kbuf, uint16_t* __restrict__ vbuf)
{
    __shared__ uint16_t sA[2][128][36];   // [m][k] bf16, 8B-aligned rows
    __shared__ uint16_t sB[2][128][36];   // [n][k] bf16, 8B-aligned rows

    const int tid   = threadIdx.x;
    const int wave  = tid >> 5;
    const int lane  = tid & 31;
    const int ln16  = lane & 15;
    const int g     = lane >> 4;
    const int waveM = (wave & 3) * 32;
    const int waveN = (wave >> 2) * 64;
    const int blockM = blockIdx.y * 128;
    const int blockN = blockIdx.x * 128;

    v8f acc[2][4];
#pragma unroll
    for (int mt = 0; mt < 2; ++mt)
#pragma unroll
        for (int nt = 0; nt < 4; ++nt)
#pragma unroll
            for (int r = 0; r < 8; ++r) acc[mt][nt][r] = 0.0f;

    GEMM_MAIN_LOOP();

    // epilogue: + bias, scatter into q/k [B,H,N,hd], v [B,H,hd,N]
#pragma unroll
    for (int mt = 0; mt < 2; ++mt) {
#pragma unroll
        for (int nt = 0; nt < 4; ++nt) {
            const int col   = blockN + waveN + nt * 16 + ln16;
            const int which = col >> 9;          // 0=q 1=k 2=v
            const int cc    = col & 511;
            const int h     = cc >> 6, d = cc & 63;
            const float bias = bqkv[col];
#pragma unroll
            for (int r = 0; r < 8; ++r) {
                const int row = blockM + waveM + mt * 16 + r + 8 * g;
                const int bb = row >> 12, n = row & 4095;
                const uint16_t bv = f32_to_bf16_bits(acc[mt][nt][r] + bias);
                if (which == 2) {
                    vbuf[(size_t)((bb * 8 + h) * 64 + d) * 4096 + n] = bv;
                } else {
                    uint16_t* dst = (which == 0) ? qb : kbuf;
                    dst[(size_t)(((bb * 8 + h) * 4096) + n) * 64 + d] = bv;
                }
            }
        }
    }
}

// ---------------------------------------------------------------------------
// Kernel 2: flash attention per (b,h). 128 threads = 4 waves, 16 q-rows/wave.
// Double-buffered async (ASYNCcnt) staging of K [32][hd] and V^T [hd][32]
// tiles; fp32 online softmax; writes [B,N,C] bf16.
// ---------------------------------------------------------------------------
__global__ __launch_bounds__(128) void attn_kernel(
    const uint16_t* __restrict__ qb, const uint16_t* __restrict__ kb,
    const uint16_t* __restrict__ vt, uint16_t* __restrict__ ob)
{
    __shared__ uint16_t sK[2][32][68];   // [key][hd] bf16, 8B-aligned rows
    __shared__ uint16_t sVt[2][64][36];  // [hd][key] bf16, 8B-aligned rows
    __shared__ uint16_t sP[4][16][34];   // per-wave P tile [m][k], padded

    const int tid  = threadIdx.x;
    const int wave = tid >> 5, lane = tid & 31;
    const int ln16 = lane & 15, g = lane >> 4;
    const int h = blockIdx.y, b = blockIdx.z;
    const int q0 = blockIdx.x * 64 + wave * 16;
    const size_t headBase = (size_t)((b * 8 + h) * 4096) * 64;   // q/k [N][hd]
    const size_t vtBase   = (size_t)((b * 8 + h) * 64) * 4096;   // v^T [hd][N]

    // Q fragments (A layout) for the two hd halves, rows q0 + (lane%16)
    v16bf qa[2];
#pragma unroll
    for (int kq = 0; kq < 2; ++kq) {
        v8u u;
#pragma unroll
        for (int v = 0; v < 8; ++v)
            u[v] = *(const uint32_t*)&qb[headBase + (size_t)(q0 + ln16) * 64 + kq * 32 + kA_idx(v, g)];
        qa[kq] = __builtin_bit_cast(v16bf, u);
    }

    float mrow[8], lrow[8];
#pragma unroll
    for (int r = 0; r < 8; ++r) { mrow[r] = -1e30f; lrow[r] = 0.0f; }
    v8f o[4];
#pragma unroll
    for (int t = 0; t < 4; ++t)
#pragma unroll
        for (int r = 0; r < 8; ++r) o[t][r] = 0.0f;

    const float inv_scale = 0.04419417382415922f;  // 1/sqrt(512)

#define STAGE_KV(bfi, key0_)                                                     \
    do {                                                                         \
        const int _k0 = (key0_);                                                 \
        _Pragma("unroll")                                                        \
        for (int _p = 0; _p < 4; ++_p) {  /* K: 512 b64 transfers / 128 thr */   \
            const int _i = _p * 128 + tid;                                       \
            const int _kr = _i >> 4, _c = _i & 15;                               \
            async_ld_b64(&sK[bfi][_kr][_c * 4],                                  \
                         &kb[headBase + (size_t)(_k0 + _kr) * 64 + _c * 4]);     \
        }                                                                        \
        _Pragma("unroll")                                                        \
        for (int _p = 0; _p < 4; ++_p) {  /* V^T: 512 b64 transfers / 128 thr */ \
            const int _i = _p * 128 + tid;                                       \
            const int _d = _i >> 3, _c = _i & 7;                                 \
            async_ld_b64(&sVt[bfi][_d][_c * 4],                                  \
                         &vt[vtBase + (size_t)_d * 4096 + _k0 + _c * 4]);        \
        }                                                                        \
    } while (0)

    STAGE_KV(0, 0);

    for (int kblk = 0; kblk < 128; ++kblk) {
        const int buf  = kblk & 1;
        const int key0 = kblk * 32;
        if (kblk + 1 < 128) {
            STAGE_KV(buf ^ 1, key0 + 32);   // prefetch next block
            ASYNC_WAIT(0x8);                // retire older batch (in-order)
        } else {
            ASYNC_WAIT(0x0);
        }
        __syncthreads();

        // S = Q K^T : two 16-key column tiles, contraction hd=64 (2 WMMA each)
        v8f s[2];
#pragma unroll
        for (int j = 0; j < 2; ++j) {
#pragma unroll
            for (int r = 0; r < 8; ++r) s[j][r] = 0.0f;
#pragma unroll
            for (int kq = 0; kq < 2; ++kq) {
                v8u u;
#pragma unroll
                for (int v = 0; v < 8; ++v)
                    u[v] = *(const uint32_t*)&sK[buf][j * 16 + ln16][kq * 32 + g * 16 + v * 2];
                v16bf kf = __builtin_bit_cast(v16bf, u);
                s[j] = __builtin_amdgcn_wmma_f32_16x16x32_bf16(
                    false, qa[kq], false, kf, (short)0, s[j], false, false);
            }
        }

        // online softmax (row m = r + 8*g lives in one 16-lane half)
#pragma unroll
        for (int r = 0; r < 8; ++r) {
            float s0 = s[0][r] * inv_scale;
            float s1 = s[1][r] * inv_scale;
            float mx = fmaxf(s0, s1);
#pragma unroll
            for (int m = 1; m < 16; m <<= 1) mx = fmaxf(mx, __shfl_xor(mx, m, 32));
            const float mnew  = fmaxf(mrow[r], mx);
            const float p0    = __expf(s0 - mnew);
            const float p1    = __expf(s1 - mnew);
            const float alpha = __expf(mrow[r] - mnew);
            float ps = p0 + p1;
#pragma unroll
            for (int m = 1; m < 16; m <<= 1) ps += __shfl_xor(ps, m, 32);
            lrow[r] = lrow[r] * alpha + ps;
            mrow[r] = mnew;
#pragma unroll
            for (int t = 0; t < 4; ++t) o[t][r] *= alpha;
            sP[wave][r + 8 * g][ln16]      = f32_to_bf16_bits(p0);
            sP[wave][r + 8 * g][16 + ln16] = f32_to_bf16_bits(p1);
        }

        // O += P @ V  (P relayout C->A via per-wave LDS; same-wave DS in-order)
        {
            v8u pu;
#pragma unroll
            for (int v = 0; v < 8; ++v)
                pu[v] = *(const uint32_t*)&sP[wave][ln16][kA_idx(v, g)];
            v16bf pf = __builtin_bit_cast(v16bf, pu);
#pragma unroll
            for (int t = 0; t < 4; ++t) {
                v8u vu;
#pragma unroll
                for (int v = 0; v < 8; ++v)
                    vu[v] = *(const uint32_t*)&sVt[buf][t * 16 + ln16][g * 16 + v * 2];
                v16bf vf = __builtin_bit_cast(v16bf, vu);
                o[t] = __builtin_amdgcn_wmma_f32_16x16x32_bf16(
                    false, pf, false, vf, (short)0, o[t], false, false);
            }
        }
        __syncthreads();   // all waves done reading buf before it is re-staged
    }
#undef STAGE_KV

    // normalize, store to attn buffer [B,N,512] bf16
#pragma unroll
    for (int t = 0; t < 4; ++t) {
        const int d = t * 16 + ln16;
#pragma unroll
        for (int r = 0; r < 8; ++r) {
            const int row = q0 + r + 8 * g;
            const float val = o[t][r] / lrow[r];
            ob[(size_t)(b * 4096 + row) * 512 + h * 64 + d] = f32_to_bf16_bits(val);
        }
    }
}

// ---------------------------------------------------------------------------
// Kernel 3: out = attn @ W_out^T(pre-transposed) + b_out.
// M = 8192, K = 512, N = 512, fp32 out.
// ---------------------------------------------------------------------------
__global__ __launch_bounds__(256) void out_gemm_kernel(
    const uint16_t* __restrict__ Ab, const uint16_t* __restrict__ Bt,
    const float* __restrict__ bout, float* __restrict__ out)
{
    __shared__ uint16_t sA[2][128][36];
    __shared__ uint16_t sB[2][128][36];

    const int tid   = threadIdx.x;
    const int wave  = tid >> 5;
    const int lane  = tid & 31;
    const int ln16  = lane & 15;
    const int g     = lane >> 4;
    const int waveM = (wave & 3) * 32;
    const int waveN = (wave >> 2) * 64;
    const int blockM = blockIdx.y * 128;
    const int blockN = blockIdx.x * 128;

    v8f acc[2][4];
#pragma unroll
    for (int mt = 0; mt < 2; ++mt)
#pragma unroll
        for (int nt = 0; nt < 4; ++nt)
#pragma unroll
            for (int r = 0; r < 8; ++r) acc[mt][nt][r] = 0.0f;

    GEMM_MAIN_LOOP();

#pragma unroll
    for (int mt = 0; mt < 2; ++mt) {
#pragma unroll
        for (int nt = 0; nt < 4; ++nt) {
            const int col = blockN + waveN + nt * 16 + ln16;
            const float bias = bout[col];
#pragma unroll
            for (int r = 0; r < 8; ++r) {
                const int row = blockM + waveM + mt * 16 + r + 8 * g;
                out[(size_t)row * 512 + col] = acc[mt][nt][r] + bias;
            }
        }
    }
}

extern "C" void kernel_launch(void* const* d_in, const int* in_sizes, int n_in,
                              void* d_out, int out_size, void* d_ws, size_t ws_size,
                              hipStream_t stream) {
    (void)in_sizes; (void)n_in; (void)out_size; (void)ws_size;
    const float* x    = (const float*)d_in[0];
    const float* Wqkv = (const float*)d_in[1];
    const float* bqkv = (const float*)d_in[2];
    const float* Wout = (const float*)d_in[3];
    const float* bout = (const float*)d_in[4];
    float* out = (float*)d_out;

    uint16_t* ws    = (uint16_t*)d_ws;
    uint16_t* qb    = ws;                               // 4M bf16 [B,H,N,hd]
    uint16_t* kbuf  = ws + (size_t)4  * 1024 * 1024;    // 4M bf16 [B,H,N,hd]
    uint16_t* vbuf  = ws + (size_t)8  * 1024 * 1024;    // 4M bf16 [B,H,hd,N]
    uint16_t* attn  = ws + (size_t)12 * 1024 * 1024;    // 4M bf16 [B,N,C]
    uint16_t* xb    = ws + (size_t)16 * 1024 * 1024;    // 4M bf16 [B*N,C]
    uint16_t* wqkvT = ws + (size_t)20 * 1024 * 1024;    // 768K bf16 [1536][512]
    uint16_t* woutT = wqkvT + 786432;                   // 256K bf16 [512][512]

    prep_kernel<<<20480, 256, 0, stream>>>(x, Wqkv, Wout, xb, wqkvT, woutT);
    qkv_gemm_kernel<<<dim3(12, 64), 256, 0, stream>>>(xb, wqkvT, bqkv, qb, kbuf, vbuf);
    attn_kernel<<<dim3(64, 8, 2), 128, 0, stream>>>(qb, kbuf, vbuf, attn);
    out_gemm_kernel<<<dim3(4, 64), 256, 0, stream>>>(attn, woutT, bout, out);
}
